// SwitchFeedForward_4535485464936
// MI455X (gfx1250) — matrix operand order repaired and checked by reading, not compiled, not run on hardware
//
#include <hip/hip_runtime.h>

typedef __bf16 bf16;
typedef bf16  bf16x16 __attribute__((ext_vector_type(16)));
typedef bf16  bf16x8  __attribute__((ext_vector_type(8)));
typedef bf16  bf16x4  __attribute__((ext_vector_type(4)));
typedef float floatx8 __attribute__((ext_vector_type(8)));
typedef float floatx4 __attribute__((ext_vector_type(4)));

#define D_MODEL 1024
#define D_FF    4096
#define NEXP    8
#define NTOK    8192
#define CAP     1280

#define BM 64
#define BN 128
#define BK 32

// CDNA5 async global->LDS copy (ASYNCcnt path). VDST = per-lane LDS byte
// address (low 32 bits of the flat LDS pointer, per ISA 10.2 aperture rules),
// VADDR = 64-bit global address, SADDR = off (GV mode).
__device__ __forceinline__ void async_copy16_to_lds(void* lds_dst, const void* gsrc) {
    unsigned lva = (unsigned)(size_t)lds_dst;
    asm volatile("global_load_async_to_lds_b128 %0, %1, off"
                 :: "v"(lva), "v"(gsrc) : "memory");
}

__device__ __forceinline__ void wait_asynccnt0() {
#if defined(__has_builtin) && __has_builtin(__builtin_amdgcn_s_wait_asynccnt)
    __builtin_amdgcn_s_wait_asynccnt(0);
#else
    asm volatile("s_wait_asynccnt 0x0" ::: "memory");
#endif
}

// ---------------------------------------------------------------- init
__global__ void init_kernel(int* cnt) {
    int i = threadIdx.x;
    if (i < NEXP) cnt[i] = 0;
}

// ---------------------------------------------------------------- gating
__global__ __launch_bounds__(256)
void gate_kernel(const float* __restrict__ x, const float* __restrict__ Wg,
                 const float* __restrict__ bg, int* __restrict__ top,
                 float* __restrict__ gate, float* __restrict__ impB,
                 int* __restrict__ cnt, int* __restrict__ hist)
{
    __shared__ float WgS[D_MODEL * NEXP];       // 32 KB
    __shared__ float red[256][NEXP];            // 8 KB
    __shared__ int   histS[NEXP];
    const int tid = threadIdx.x;

    for (int i = tid; i < D_MODEL * NEXP; i += 256) WgS[i] = Wg[i];
    if (tid < NEXP) histS[tid] = 0;
    __syncthreads();

    const int t = blockIdx.x * 256 + tid;
    float acc[NEXP];
#pragma unroll
    for (int e = 0; e < NEXP; ++e) acc[e] = bg[e];

    const floatx4* xin = (const floatx4*)(x + (size_t)t * D_MODEL);
    for (int i = 0; i < D_MODEL / 4; ++i) {
        floatx4 v = xin[i];
#pragma unroll
        for (int q = 0; q < 4; ++q) {
            float xv = v[q];
            const float* w = &WgS[(i * 4 + q) * NEXP];
#pragma unroll
            for (int e = 0; e < NEXP; ++e) acc[e] += xv * w[e];
        }
    }

    // softmax + argmax (first-max tie-break, like jnp.argmax)
    float m = acc[0]; int best = 0;
#pragma unroll
    for (int e = 1; e < NEXP; ++e) if (acc[e] > m) { m = acc[e]; best = e; }
    float p[NEXP], s = 0.f;
#pragma unroll
    for (int e = 0; e < NEXP; ++e) { p[e] = __expf(acc[e] - m); s += p[e]; }
    float inv = 1.f / s;

    top[t]  = best;
    gate[t] = p[best] * inv;
    atomicAdd(&histS[best], 1);                  // integer: deterministic

#pragma unroll
    for (int e = 0; e < NEXP; ++e) red[tid][e] = p[e] * inv;
    __syncthreads();
    for (int st = 128; st > 0; st >>= 1) {       // fixed-order tree: deterministic
        if (tid < st) {
#pragma unroll
            for (int e = 0; e < NEXP; ++e) red[tid][e] += red[tid + st][e];
        }
        __syncthreads();
    }
    if (tid < NEXP) {
        impB[blockIdx.x * NEXP + tid] = red[0][tid];
        hist[blockIdx.x * NEXP + tid] = histS[tid];
        atomicAdd(&cnt[tid], histS[tid]);
    }
}

// ---------------------------------------------------------------- prefix over block hists
__global__ void scan_kernel(const int* __restrict__ hist, int* __restrict__ base) {
    int e = threadIdx.x;
    if (e < NEXP) {
        int run = 0;
        for (int b = 0; b < NTOK / 256; ++b) {
            base[b * NEXP + e] = run;
            run += hist[b * NEXP + e];
        }
    }
}

// ---------------------------------------------------------------- per-token rank/slot
__global__ __launch_bounds__(256)
void rank_kernel(const int* __restrict__ top, const int* __restrict__ base,
                 int* __restrict__ idx)
{
    __shared__ int topS[256];
    const int tid = threadIdx.x;
    const int t = blockIdx.x * 256 + tid;
    const int e = top[t];
    topS[tid] = e;
    __syncthreads();
    int local = 0;
    for (int j = 0; j < tid; ++j) local += (topS[j] == e) ? 1 : 0;
    int rank = base[blockIdx.x * NEXP + e] + local;
    idx[t] = (rank < CAP) ? (e * CAP + rank) : -1;
}

// ---------------------------------------------------------------- dispatch x -> be (bf16)
__global__ __launch_bounds__(256)
void dispatch_kernel(const float* __restrict__ x, const int* __restrict__ idx,
                     bf16* __restrict__ be)
{
    const int t = blockIdx.x;
    const int s = idx[t];
    if (s < 0) return;
    const int tid = threadIdx.x;
    floatx4 v = *(const floatx4*)(x + (size_t)t * D_MODEL + tid * 4);
    bf16x4 o = { (bf16)v[0], (bf16)v[1], (bf16)v[2], (bf16)v[3] };
    *(bf16x4*)(be + (size_t)s * D_MODEL + tid * 4) = o;
}

// ---------------------------------------------------------------- WMMA GEMM
// C[M,N] = A[M,K](bf16) * B[K,N](f32, converted to bf16 in LDS) + bias[N]
template <bool RELU, bool OUT_BF16>
__global__ __launch_bounds__(256)
void gemm_kernel(const bf16* __restrict__ A, const float* __restrict__ Bw,
                 const float* __restrict__ bias, void* __restrict__ outp,
                 int M, int N, int K)
{
    __shared__ alignas(64) bf16 As[BM][BK];     // 4 KB
    __shared__ alignas(64) bf16 BsT[BN][BK];    // 8 KB, transposed [n][k]

    const int tid   = threadIdx.x;
    const int lane  = tid & 31;
    const int wave  = tid >> 5;
    const int waveM = wave >> 2;                 // 0..1
    const int waveN = wave & 3;                  // 0..3
    const int rowBase = blockIdx.y * BM;
    const int colBase = blockIdx.x * BN;
    const int half = lane >> 4;
    const int l15  = lane & 15;

    floatx8 acc[2][2] = {};

    // per-thread A-tile copy coordinates (one 16B chunk per thread)
    const int ar = tid >> 2;
    const int ac = (tid & 3) * 8;
    // per-thread B-tile micro-tile: 4(k) x 4(n)
    const int bk0 = (tid >> 5) * 4;              // 0,4,...,28
    const int bn0 = (tid & 31) * 4;              // 0,4,...,124

    for (int kb = 0; kb < K; kb += BK) {
        // ---- A tile: 64x32 bf16 (pure move -> async LDS DMA on CDNA5)
        async_copy16_to_lds(&As[ar][ac], &A[(size_t)(rowBase + ar) * K + kb + ac]);

        // ---- B tile: 32x128 f32 -> bf16, stored transposed [n][k]
        {
            const float* src = &Bw[(size_t)(kb + bk0) * N + colBase + bn0];
            floatx4 r0 = *(const floatx4*)(src);
            floatx4 r1 = *(const floatx4*)(src + N);
            floatx4 r2 = *(const floatx4*)(src + 2 * (size_t)N);
            floatx4 r3 = *(const floatx4*)(src + 3 * (size_t)N);
#pragma unroll
            for (int n = 0; n < 4; ++n) {
                bf16x4 p = { (bf16)r0[n], (bf16)r1[n], (bf16)r2[n], (bf16)r3[n] };
                *(bf16x4*)&BsT[bn0 + n][bk0] = p;   // one ds_store_b64 per n
            }
        }
        wait_asynccnt0();
        __syncthreads();

        // ---- fragments per ISA 7.12.2 layouts
        bf16x16 afrag[2], bfrag[2];
#pragma unroll
        for (int tm = 0; tm < 2; ++tm) {
            int row = waveM * 32 + tm * 16 + l15;
            bf16x8 lo = *(const bf16x8*)&As[row][half * 8];        // K = 8h .. 8h+7
            bf16x8 hi = *(const bf16x8*)&As[row][16 + half * 8];   // K = 16+8h ..
#pragma unroll
            for (int j = 0; j < 8; ++j) {
                afrag[tm][j]     = lo[j];
                afrag[tm][8 + j] = hi[j];
            }
        }
#pragma unroll
        for (int tn = 0; tn < 2; ++tn) {
            int col = waveN * 32 + tn * 16 + l15;
            bf16x8 lo = *(const bf16x8*)&BsT[col][half * 8];       // K = 8h .. 8h+7
            bf16x8 hi = *(const bf16x8*)&BsT[col][16 + half * 8];  // K = 16+8h ..
#pragma unroll
            for (int j = 0; j < 8; ++j) {
                bfrag[tn][j]     = lo[j];
                bfrag[tn][8 + j] = hi[j];
            }
        }

#pragma unroll
        for (int tm = 0; tm < 2; ++tm)
#pragma unroll
            for (int tn = 0; tn < 2; ++tn)
                acc[tm][tn] = __builtin_amdgcn_wmma_f32_16x16x32_bf16(
                    false, afrag[tm], false, bfrag[tn],
                    (short)0, acc[tm][tn], false, false);
        __syncthreads();
    }

    // ---- epilogue: C/D layout -> row = 8*(lane>>4)+r, col = lane&15
#pragma unroll
    for (int tm = 0; tm < 2; ++tm) {
#pragma unroll
        for (int tn = 0; tn < 2; ++tn) {
            int tileRow = rowBase + waveM * 32 + tm * 16;
            int col = colBase + waveN * 32 + tn * 16 + l15;
            float b = bias[col];
#pragma unroll
            for (int r = 0; r < 8; ++r) {
                int row = tileRow + half * 8 + r;
                float v = acc[tm][tn][r] + b;
                if (RELU) v = v > 0.f ? v : 0.f;
                if (OUT_BF16)
                    ((bf16*)outp)[(size_t)row * N + col] = (bf16)v;
                else
                    ((float*)outp)[(size_t)row * N + col] = v;
            }
        }
    }
}

// ---------------------------------------------------------------- gather + gate scale
__global__ __launch_bounds__(256)
void gather_kernel(const float* __restrict__ O, const int* __restrict__ idx,
                   const float* __restrict__ gate, float* __restrict__ y)
{
    const int t = blockIdx.x;
    const int tid = threadIdx.x;
    const int s = idx[t];
    floatx4 o = {0.f, 0.f, 0.f, 0.f};
    if (s >= 0) {
        float g = gate[t];
        floatx4 v = *(const floatx4*)(O + (size_t)s * D_MODEL + tid * 4);
        o = v * g;
    }
    *(floatx4*)(y + (size_t)t * D_MODEL + tid * 4) = o;
}

// ---------------------------------------------------------------- aux loss
__global__ void aux_kernel(const float* __restrict__ impB, const int* __restrict__ cnt,
                           float* __restrict__ out)
{
    if (threadIdx.x == 0) {
        float s = 0.f;
        for (int e = 0; e < NEXP; ++e) {
            float imp = 0.f;
            for (int b = 0; b < NTOK / 256; ++b) imp += impB[b * NEXP + e];
            s += (imp / (float)NTOK) * ((float)cnt[e] / (float)NTOK);
        }
        out[0] = s * (float)NEXP;
    }
}

// ---------------------------------------------------------------- host
extern "C" void kernel_launch(void* const* d_in, const int* in_sizes, int n_in,
                              void* d_out, int out_size, void* d_ws, size_t ws_size,
                              hipStream_t stream)
{
    const float* x  = (const float*)d_in[0];
    const float* Wg = (const float*)d_in[1];
    const float* bg = (const float*)d_in[2];
    const float* W1 = (const float*)d_in[3];
    const float* b1 = (const float*)d_in[4];
    const float* W2 = (const float*)d_in[5];
    const float* b2 = (const float*)d_in[6];
    float* y = (float*)d_out;

    char* ws = (char*)d_ws;
    float* impB = (float*)(ws + 0);            // 32*8 f32
    int*   cnt  = (int*)(ws + 1024);           // 8 i32
    int*   hist = (int*)(ws + 2048);           // 32*8 i32
    int*   base = (int*)(ws + 3072);           // 32*8 i32
    int*   top  = (int*)(ws + 4096);           // 8192 i32
    float* gate = (float*)(ws + 36864);        // 8192 f32
    int*   idx  = (int*)(ws + 69632);          // 8192 i32
    bf16*  be   = (bf16*)(ws + 131072);        // 8*1280*1024 bf16 = 20 MiB
    bf16*  H    = (bf16*)(ws + 21102592ull);   // 1280*4096 bf16  = 10 MiB (reused per expert)
    float* O    = (float*)(ws + 31588352ull);  // 8*1280*1024 f32 = 40 MiB

    init_kernel<<<1, 32, 0, stream>>>(cnt);
    gate_kernel<<<NTOK / 256, 256, 0, stream>>>(x, Wg, bg, top, gate, impB, cnt, hist);
    scan_kernel<<<1, 32, 0, stream>>>(hist, base);
    rank_kernel<<<NTOK / 256, 256, 0, stream>>>(top, base, idx);
    dispatch_kernel<<<NTOK, 256, 0, stream>>>(x, idx, be);

    for (int e = 0; e < NEXP; ++e) {
        gemm_kernel<true, true><<<dim3(D_FF / BN, CAP / BM), 256, 0, stream>>>(
            be + (size_t)e * CAP * D_MODEL,
            W1 + (size_t)e * D_MODEL * D_FF,
            b1 + (size_t)e * D_FF,
            (void*)H, CAP, D_FF, D_MODEL);
        gemm_kernel<false, false><<<dim3(D_MODEL / BN, CAP / BM), 256, 0, stream>>>(
            H,
            W2 + (size_t)e * D_FF * D_MODEL,
            b2 + (size_t)e * D_MODEL,
            (void*)(O + (size_t)e * CAP * D_MODEL), CAP, D_MODEL, D_FF);
    }

    gather_kernel<<<NTOK, 256, 0, stream>>>(O, idx, gate, y);
    aux_kernel<<<1, 32, 0, stream>>>(impB, cnt, y + (size_t)NTOK * D_MODEL);
}
